// HELMEncoder_335007449221
// MI455X (gfx1250) — matrix operand order repaired and checked by reading, not compiled, not run on hardware
//
#include <hip/hip_runtime.h>

#define BB 64
#define DD 1024
#define VV 267735
#define VP 268288           // 64*4192 = 1024*262, padded vocab
#define NB6 262             // VP/1024 partial blocks for state GEMM
#define NHH 16
#define DHH 64
#define KLEN 512
#define DI 4096
#define NL 4
#define MEMN 511
#define IND 7056
#define BETA 100.0f
#define SPAD 1042           // odd-dword pitch (bf16) for sim-kernel LDS tiles
#define TPAD 34             // pitch for 32-wide transposed tiles

typedef __attribute__((ext_vector_type(16))) __bf16 v16bf;
typedef __attribute__((ext_vector_type(8))) float v8f;

__device__ __forceinline__ __bf16 f2bf(float f) {
  unsigned u = __builtin_bit_cast(unsigned, f);
  unsigned r = u + 0x7FFFu + ((u >> 16) & 1u);
  unsigned short h = (unsigned short)(r >> 16);
  return __builtin_bit_cast(__bf16, h);
}
__device__ __forceinline__ float bf2f(__bf16 b) {
  unsigned short h = __builtin_bit_cast(unsigned short, b);
  unsigned u = ((unsigned)h) << 16;
  return __builtin_bit_cast(float, u);
}

// A fragment (16x32 bf16): lane = M (mod 16); lanes 0-15 hold K {0..7,16..23},
// lanes 16-31 hold K {8..15,24..31}. Source: row-major [M][>=32] LDS tile.
__device__ __forceinline__ v16bf load_a_frag(const __bf16* base, int stride, int m0, int lane) {
  const __bf16* p = base + (size_t)(m0 + (lane & 15)) * stride + ((lane & 16) ? 8 : 0);
  v16bf f;
#pragma unroll
  for (int i = 0; i < 8; ++i) { f[i] = p[i]; f[i + 8] = p[16 + i]; }
  return f;
}
// B fragment (32x16 bf16): lane = N (mod 16); lanes 0-15 hold K {0..7,16..23},
// lanes 16-31 hold K {8..15,24..31}. Source: N-major transposed LDS tile [N][>=32].
__device__ __forceinline__ v16bf load_b_frag(const __bf16* base, int stride, int n0, int lane) {
  const __bf16* p = base + (size_t)(n0 + (lane & 15)) * stride + ((lane & 16) ? 8 : 0);
  v16bf f;
#pragma unroll
  for (int i = 0; i < 8; ++i) { f[i] = p[i]; f[i + 8] = p[16 + i]; }
  return f;
}
#define WMMA_BF16(a, b, c) \
  __builtin_amdgcn_wmma_f32_16x16x32_bf16(false, (a), false, (b), (short)0, (c), false, false)

// ---------------- obs mean over channels ----------------
__global__ void k_mean(const float* __restrict__ obs, float* __restrict__ x) {
  int b = blockIdx.x;
  const float* o0 = obs + (size_t)b * 3 * IND;
  for (int i = threadIdx.x; i < IND; i += blockDim.x)
    x[(size_t)b * IND + i] = (o0[i] + o0[IND + i] + o0[2 * IND + i]) * (1.0f / 3.0f);
}

// ---------------- p = x @ rand_proj^T (64x7056 * 7056x1024) ----------------
#define PCH 64
__global__ void k_proj(const float* __restrict__ x, const float* __restrict__ rp,
                       float* __restrict__ p) {
  __shared__ float xs[64][PCH];
  __shared__ float ww[64][PCH];
  int d0 = blockIdx.x * 64;
  int t = threadIdx.x;
  int dloc = t & 63, bg = t >> 6;  // 4 b-groups of 16
  float acc[16];
#pragma unroll
  for (int i = 0; i < 16; ++i) acc[i] = 0.f;
  for (int k0 = 0; k0 < IND; k0 += PCH) {
    __syncthreads();
    for (int idx = t; idx < 64 * PCH; idx += 256) {
      int r = idx / PCH, c = idx % PCH;
      int k = k0 + c;
      xs[r][c] = (k < IND) ? x[(size_t)r * IND + k] : 0.f;
      ww[r][c] = (k < IND) ? rp[(size_t)(d0 + r) * IND + k] : 0.f;
    }
    __syncthreads();
    for (int c = 0; c < PCH; ++c) {
      float wv = ww[dloc][c];
#pragma unroll
      for (int i = 0; i < 16; ++i) acc[i] += wv * xs[bg * 16 + i][c];
    }
  }
#pragma unroll
  for (int i = 0; i < 16; ++i) p[(size_t)(bg * 16 + i) * DD + d0 + dloc] = acc[i];
}

__global__ void k_rownorm(const float* __restrict__ p, float* __restrict__ pn) {
  __shared__ float red[256];
  int b = blockIdx.x, t = threadIdx.x;
  float s = 0;
  for (int i = t; i < DD; i += 256) { float v = p[(size_t)b * DD + i]; s += v * v; }
  red[t] = s; __syncthreads();
  for (int w = 128; w > 0; w >>= 1) { if (t < w) red[t] += red[t + w]; __syncthreads(); }
  if (t == 0) pn[b] = sqrtf(red[0]);
}

// ---------------- sim = BETA * (p @ E^T) / (|p||e|+eps), WMMA bf16 ----------------
__global__ void k_sim(const float* __restrict__ p, const float* __restrict__ pn,
                      const float* __restrict__ E, float* __restrict__ sim) {
  extern __shared__ unsigned char smem[];
  __bf16* pl = (__bf16*)smem;               // [64][SPAD]
  __bf16* el = pl + 64 * SPAD;              // [64][SPAD]
  float* en = (float*)(el + 64 * SPAD);     // [64]
  float* red = en + 64;                     // [512]
  int t = threadIdx.x;
  int v0 = blockIdx.x * 64;
  for (int idx = t; idx < 64 * 1024; idx += 512) {
    int r = idx >> 10, c = idx & 1023;
    pl[r * SPAD + c] = f2bf(p[(size_t)r * 1024 + c]);
    int v = v0 + r;
    float ev = (v < VV) ? E[(size_t)v * 1024 + c] : 0.f;
    el[r * SPAD + c] = f2bf(ev);
  }
  __syncthreads();
  {
    int r = t >> 3, seg = t & 7;
    float s = 0;
    for (int c = seg * 128; c < seg * 128 + 128; ++c) { float v = bf2f(el[r * SPAD + c]); s += v * v; }
    red[t] = s;
  }
  __syncthreads();
  if (t < 64) {
    float s = 0;
    for (int i = 0; i < 8; ++i) s += red[t * 8 + i];
    en[t] = sqrtf(s);
  }
  __syncthreads();
  int lane = t & 31, wv = t >> 5;  // 16 waves
  int bg = wv & 3, vg = wv >> 2;   // 4 x 4
  v8f c; for (int i = 0; i < 8; ++i) c[i] = 0.f;
  int m0 = bg * 16, n0 = vg * 16;
  for (int k0 = 0; k0 < 1024; k0 += 32) {
    v16bf a = load_a_frag(pl + k0, SPAD, m0, lane);
    v16bf bfr = load_b_frag(el + k0, SPAD, n0, lane);
    c = WMMA_BF16(a, bfr, c);
  }
  int n = n0 + (lane & 15);
  float enl = en[n];
  size_t vglob = (size_t)v0 + n;
#pragma unroll
  for (int r = 0; r < 8; ++r) {
    int m = m0 + r + ((lane & 16) ? 8 : 0);
    sim[(size_t)m * VP + vglob] =
        c[r] * BETA * __builtin_amdgcn_rcpf(pn[m] * enl + 1e-8f);
  }
}

__global__ void k_rowmax(const float* __restrict__ sim, float* __restrict__ m) {
  __shared__ float red[1024];
  int b = blockIdx.x, t = threadIdx.x;
  float mx = -1e30f;
  const float* row = sim + (size_t)b * VP;
  for (int v = t; v < VV; v += 1024) mx = fmaxf(mx, row[v]);
  red[t] = mx; __syncthreads();
  for (int w = 512; w > 0; w >>= 1) { if (t < w) red[t] = fmaxf(red[t], red[t + w]); __syncthreads(); }
  if (t == 0) m[b] = red[0];
}

__global__ void k_exp(const float* __restrict__ sim, const float* __restrict__ m,
                      __bf16* __restrict__ probs, float* __restrict__ s) {
  __shared__ float red[1024];
  int b = blockIdx.x, t = threadIdx.x;
  float mx = m[b], acc = 0;
  const float* row = sim + (size_t)b * VP;
  __bf16* prow = probs + (size_t)b * VP;
  for (int v = t; v < VP; v += 1024) {
    float e = (v < VV) ? __expf(row[v] - mx) : 0.f;
    prow[v] = f2bf(e);
    acc += e;
  }
  red[t] = acc; __syncthreads();
  for (int w = 512; w > 0; w >>= 1) { if (t < w) red[t] += red[t + w]; __syncthreads(); }
  if (t == 0) s[b] = red[0];
}

// ---------------- partial state = probs_chunk @ E_chunk, WMMA bf16 ----------------
__global__ void k_av(const __bf16* __restrict__ probs, const float* __restrict__ E,
                     float* __restrict__ part) {
  extern __shared__ unsigned char smem[];
  __bf16* et = (__bf16*)smem;  // [1024][TPAD] transposed E tile
  int t = threadIdx.x, lane = t & 31, wv = t >> 5;  // 32 waves
  int bg = wv & 3, dg = wv >> 2;                    // 4 x 8 (dg -> 128 d)
  int vbase = blockIdx.x * 1024;
  v8f acc[8];
#pragma unroll
  for (int i = 0; i < 8; ++i)
    for (int r = 0; r < 8; ++r) acc[i][r] = 0.f;
  for (int v0 = vbase; v0 < vbase + 1024; v0 += 32) {
    __syncthreads();
    for (int idx = t; idx < 32 * 1024; idx += 1024) {
      int vi = idx >> 10, d = idx & 1023;
      int v = v0 + vi;
      float ev = (v < VV) ? E[(size_t)v * 1024 + d] : 0.f;
      et[d * TPAD + vi] = f2bf(ev);
      // keep L2 one tile ahead of the streaming consume (global_prefetch_b8)
      int vn = v + 32;
      if (vn < VV) __builtin_prefetch(E + (size_t)vn * 1024 + d, 0, 1);
    }
    __syncthreads();
    v16bf a;
    {
      int m = bg * 16 + (lane & 15);
      const __bf16* pr = probs + (size_t)m * VP + v0 + ((lane & 16) ? 8 : 0);
#pragma unroll
      for (int i = 0; i < 8; ++i) { a[i] = pr[i]; a[i + 8] = pr[16 + i]; }
    }
#pragma unroll
    for (int tl = 0; tl < 8; ++tl) {
      v16bf bf = load_b_frag(et, TPAD, dg * 128 + tl * 16, lane);
      acc[tl] = WMMA_BF16(a, bf, acc[tl]);
    }
  }
  float* po = part + (size_t)blockIdx.x * 64 * 1024;
#pragma unroll
  for (int tl = 0; tl < 8; ++tl) {
    int n = dg * 128 + tl * 16 + (lane & 15);
#pragma unroll
    for (int r = 0; r < 8; ++r) {
      int m = bg * 16 + r + ((lane & 16) ? 8 : 0);
      po[(size_t)m * 1024 + n] = acc[tl][r];
    }
  }
}

__global__ void k_merge(const float* __restrict__ part, const float* __restrict__ s,
                        float* __restrict__ h) {
  int b = blockIdx.x, d = threadIdx.x;  // 1024 threads
  float acc = 0;
  for (int blk = 0; blk < NB6; ++blk) acc += part[((size_t)blk * 64 + b) * 1024 + d];
  h[(size_t)b * 1024 + d] = acc / s[b];
}

// ---------------- generic WMMA GEMM: C = op(A@B), fp32 in, fp32 out ----------------
// Register-staged double buffering: global loads for tile k+1 are in flight while
// WMMAs consume tile k from LDS. mode: 0 plain, 1 +bias, 2 +bias,relu.
// grid.x = N/128, grid.y = ceil(M/128), 512 threads.
__global__ void k_gemm(const float* __restrict__ A, int lda,
                       const float* __restrict__ Bm, int ldb, int bcol,
                       float* __restrict__ C, int ldc, int M, int K,
                       const float* __restrict__ bias, int mode) {
  __shared__ __bf16 al[2][128 * TPAD];
  __shared__ __bf16 bt[2][128 * TPAD];
  int t = threadIdx.x, lane = t & 31, wv = t >> 5;  // 512 thr, 16 waves
  int wm = wv & 3, wn = wv >> 2;                    // 4x4 grid of 32x32 wave tiles
  int n0b = blockIdx.x * 128, m0b = blockIdx.y * 128;
  int arow = t >> 5, acol = t & 31;                 // A stage: 8 rows stride 16
  int brow = t >> 7, bcl = t & 127;                 // B stage: 8 rows stride 4
  float ra[8], rb[8];
  v8f acc[2][2];
#pragma unroll
  for (int i = 0; i < 2; ++i)
    for (int j = 0; j < 2; ++j)
      for (int r = 0; r < 8; ++r) acc[i][j][r] = 0.f;

  auto gload = [&](int k0) {
#pragma unroll
    for (int s = 0; s < 8; ++s) {
      int m = m0b + arow + 16 * s;
      ra[s] = (m < M) ? A[(size_t)m * lda + k0 + acol] : 0.f;
      rb[s] = Bm[(size_t)(k0 + brow + 4 * s) * ldb + bcol + n0b + bcl];
    }
  };
  auto lstore = [&](int buf) {
#pragma unroll
    for (int s = 0; s < 8; ++s) {
      al[buf][(arow + 16 * s) * TPAD + acol] = f2bf(ra[s]);
      bt[buf][bcl * TPAD + brow + 4 * s] = f2bf(rb[s]);
    }
  };

  gload(0);
  lstore(0);
  __syncthreads();
  int nk = K >> 5;
  for (int ki = 0; ki < nk; ++ki) {
    int cur = ki & 1;
    if (ki + 1 < nk) gload((ki + 1) << 5);  // overlap with WMMA below
    v16bf a0 = load_a_frag(al[cur], TPAD, wm * 32, lane);
    v16bf a1 = load_a_frag(al[cur], TPAD, wm * 32 + 16, lane);
    v16bf b0 = load_b_frag(bt[cur], TPAD, wn * 32, lane);
    v16bf b1 = load_b_frag(bt[cur], TPAD, wn * 32 + 16, lane);
    acc[0][0] = WMMA_BF16(a0, b0, acc[0][0]);
    acc[0][1] = WMMA_BF16(a0, b1, acc[0][1]);
    acc[1][0] = WMMA_BF16(a1, b0, acc[1][0]);
    acc[1][1] = WMMA_BF16(a1, b1, acc[1][1]);
    __syncthreads();
    if (ki + 1 < nk) lstore(cur ^ 1);
    __syncthreads();
  }
#pragma unroll
  for (int mi = 0; mi < 2; ++mi)
#pragma unroll
    for (int ni = 0; ni < 2; ++ni) {
      int n = n0b + wn * 32 + ni * 16 + (lane & 15);
#pragma unroll
      for (int r = 0; r < 8; ++r) {
        int m = m0b + wm * 32 + mi * 16 + r + ((lane & 16) ? 8 : 0);
        if (m < M) {
          float v = acc[mi][ni][r];
          if (mode >= 1) v += bias[n];
          if (mode == 2) v = fmaxf(v, 0.f);
          C[(size_t)m * ldc + n] = v;
        }
      }
    }
}

// ---------------- K/V GEMM over cat=[mems[l]; h], bf16 output ----------------
// M = 512*64 rows (row j: pos=j>>6, b=j&63). grid.x = N/128 (=16), grid.y = 256.
__global__ void k_gemm_cat(const float* __restrict__ mems_l, const float* __restrict__ h,
                           const float* __restrict__ Bm, int ldb, int bcol,
                           __bf16* __restrict__ C, int ldc, int K) {
  __shared__ __bf16 al[2][128 * TPAD];
  __shared__ __bf16 bt[2][128 * TPAD];
  int t = threadIdx.x, lane = t & 31, wv = t >> 5;
  int wm = wv & 3, wn = wv >> 2;
  int n0b = blockIdx.x * 128, m0b = blockIdx.y * 128;
  int arow = t >> 5, acol = t & 31;
  int brow = t >> 7, bcl = t & 127;
  float ra[8], rb[8];
  v8f acc[2][2];
#pragma unroll
  for (int i = 0; i < 2; ++i)
    for (int j = 0; j < 2; ++j)
      for (int r = 0; r < 8; ++r) acc[i][j][r] = 0.f;

  auto gload = [&](int k0) {
#pragma unroll
    for (int s = 0; s < 8; ++s) {
      int j = m0b + arow + 16 * s;
      int pos = j >> 6, b = j & 63;
      const float* src = (pos < MEMN) ? (mems_l + ((size_t)pos * 64 + b) * 1024)
                                      : (h + (size_t)b * 1024);
      ra[s] = src[k0 + acol];
      rb[s] = Bm[(size_t)(k0 + brow + 4 * s) * ldb + bcol + n0b + bcl];
    }
  };
  auto lstore = [&](int buf) {
#pragma unroll
    for (int s = 0; s < 8; ++s) {
      al[buf][(arow + 16 * s) * TPAD + acol] = f2bf(ra[s]);
      bt[buf][bcl * TPAD + brow + 4 * s] = f2bf(rb[s]);
    }
  };

  gload(0);
  lstore(0);
  __syncthreads();
  int nk = K >> 5;
  for (int ki = 0; ki < nk; ++ki) {
    int cur = ki & 1;
    if (ki + 1 < nk) gload((ki + 1) << 5);
    v16bf a0 = load_a_frag(al[cur], TPAD, wm * 32, lane);
    v16bf a1 = load_a_frag(al[cur], TPAD, wm * 32 + 16, lane);
    v16bf b0 = load_b_frag(bt[cur], TPAD, wn * 32, lane);
    v16bf b1 = load_b_frag(bt[cur], TPAD, wn * 32 + 16, lane);
    acc[0][0] = WMMA_BF16(a0, b0, acc[0][0]);
    acc[0][1] = WMMA_BF16(a0, b1, acc[0][1]);
    acc[1][0] = WMMA_BF16(a1, b0, acc[1][0]);
    acc[1][1] = WMMA_BF16(a1, b1, acc[1][1]);
    __syncthreads();
    if (ki + 1 < nk) lstore(cur ^ 1);
    __syncthreads();
  }
#pragma unroll
  for (int mi = 0; mi < 2; ++mi)
#pragma unroll
    for (int ni = 0; ni < 2; ++ni) {
      int n = n0b + wn * 32 + ni * 16 + (lane & 15);
#pragma unroll
      for (int r = 0; r < 8; ++r) {
        int m = m0b + wm * 32 + mi * 16 + r + ((lane & 16) ? 8 : 0);
        C[(size_t)m * ldc + n] = f2bf(acc[mi][ni][r]);
      }
    }
}

// ---------------- sinusoidal relative position embeddings ----------------
__global__ void k_posemb(float* __restrict__ r) {
  int j = blockIdx.x;  // 512
  float pos = (float)(KLEN - 1 - j);
  for (int i = threadIdx.x; i < 512; i += blockDim.x) {
    float inv = __powf(10000.f, -(float)(2 * i) / 1024.f);
    float ang = pos * inv;
    r[(size_t)j * 1024 + i] = __sinf(ang);
    r[(size_t)j * 1024 + 512 + i] = __cosf(ang);
  }
}

// ---------------- relative attention (qlen=1): scores + softmax + AV ----------------
__global__ void k_attn(const float* __restrict__ q, const __bf16* __restrict__ kv,
                       const float* __restrict__ rk,
                       const float* __restrict__ rwb, const float* __restrict__ rrb,
                       float* __restrict__ av) {
  __shared__ float qa[64], qb[64], sc[512], red[256], redv[4][64];
  int b = blockIdx.x, hh = blockIdx.y, t = threadIdx.x;
  if (t < 64) {
    float qv = q[(size_t)b * 1024 + hh * 64 + t];
    qa[t] = qv + rwb[hh * 64 + t];
    qb[t] = qv + rrb[hh * 64 + t];
  }
  __syncthreads();
  for (int j = t; j < 512; j += 256) {
    const __bf16* kr = kv + ((size_t)j * 64 + b) * 2048 + hh * 64;
    const float* rr = rk + (size_t)j * 1024 + hh * 64;
    float d = 0;
    for (int i = 0; i < 64; ++i) d += qa[i] * bf2f(kr[i]) + qb[i] * rr[i];
    sc[j] = d * 0.125f;
  }
  __syncthreads();
  red[t] = fmaxf(sc[t], sc[t + 256]); __syncthreads();
  for (int w = 128; w > 0; w >>= 1) { if (t < w) red[t] = fmaxf(red[t], red[t + w]); __syncthreads(); }
  float mx = red[0];
  __syncthreads();
  float e0 = __expf(sc[t] - mx), e1 = __expf(sc[t + 256] - mx);
  sc[t] = e0; sc[t + 256] = e1;
  red[t] = e0 + e1; __syncthreads();
  for (int w = 128; w > 0; w >>= 1) { if (t < w) red[t] += red[t + w]; __syncthreads(); }
  float inv = 1.0f / red[0];
  __syncthreads();
  int d = t & 63, jg = t >> 6;
  float a = 0;
  for (int j = jg * 128; j < jg * 128 + 128; ++j) {
    const __bf16* vr = kv + ((size_t)j * 64 + b) * 2048 + 1024 + hh * 64;
    a += sc[j] * bf2f(vr[d]);
  }
  redv[jg][d] = a; __syncthreads();
  if (t < 64)
    av[(size_t)b * 1024 + hh * 64 + t] =
        (redv[0][t] + redv[1][t] + redv[2][t] + redv[3][t]) * inv;
}

// ---------------- out = LayerNorm(a + b) * g + be ----------------
__global__ void k_addln(const float* __restrict__ a, const float* __restrict__ bb,
                        const float* __restrict__ g, const float* __restrict__ be,
                        float* __restrict__ out) {
  __shared__ float red[256];
  __shared__ float row[1024];
  int b = blockIdx.x, t = threadIdx.x;
  float s = 0;
  for (int i = t; i < 1024; i += 256) {
    float v = a[(size_t)b * 1024 + i] + bb[(size_t)b * 1024 + i];
    row[i] = v; s += v;
  }
  red[t] = s; __syncthreads();
  for (int w = 128; w > 0; w >>= 1) { if (t < w) red[t] += red[t + w]; __syncthreads(); }
  float mean = red[0] * (1.0f / 1024.0f);
  __syncthreads();
  float vv = 0;
  for (int i = t; i < 1024; i += 256) { float d = row[i] - mean; vv += d * d; }
  red[t] = vv; __syncthreads();
  for (int w = 128; w > 0; w >>= 1) { if (t < w) red[t] += red[t + w]; __syncthreads(); }
  float rstd = rsqrtf(red[0] * (1.0f / 1024.0f) + 1e-5f);
  __syncthreads();
  for (int i = t; i < 1024; i += 256)
    out[(size_t)b * 1024 + i] = (row[i] - mean) * rstd * g[i] + be[i];
}

extern "C" void kernel_launch(void* const* d_in, const int* in_sizes, int n_in,
                              void* d_out, int out_size, void* d_ws, size_t ws_size,
                              hipStream_t stream) {
  (void)in_sizes; (void)n_in; (void)out_size; (void)ws_size;
  const float* obs       = (const float*)d_in[0];
  const float* word_embs = (const float*)d_in[1];
  const float* rand_proj = (const float*)d_in[2];
  const float* mems      = (const float*)d_in[3];
  const float* qkv_w     = (const float*)d_in[4];
  const float* o_w       = (const float*)d_in[5];
  const float* r_w       = (const float*)d_in[6];
  const float* r_w_bias  = (const float*)d_in[7];
  const float* r_r_bias  = (const float*)d_in[8];
  const float* ln1_s     = (const float*)d_in[9];
  const float* ln1_b     = (const float*)d_in[10];
  const float* ff_w1     = (const float*)d_in[11];
  const float* ff_b1     = (const float*)d_in[12];
  const float* ff_w2     = (const float*)d_in[13];
  const float* ff_b2     = (const float*)d_in[14];
  const float* ln2_s     = (const float*)d_in[15];
  const float* ln2_b     = (const float*)d_in[16];
  float* out = (float*)d_out;

  unsigned char* w = (unsigned char*)d_ws;
  size_t off = 0;
  auto take = [&](size_t bytes) -> void* {
    void* p = (void*)(w + off);
    off += (bytes + 511) & ~(size_t)511;
    return p;
  };
  float*  xw    = (float*)take((size_t)64 * IND * 4);
  float*  pw    = (float*)take((size_t)64 * 1024 * 4);
  float*  pn    = (float*)take(64 * 4);
  float*  simw  = (float*)take((size_t)64 * VP * 4);
  float*  mrow  = (float*)take(64 * 4);
  float*  srow  = (float*)take(64 * 4);
  __bf16* probs = (__bf16*)take((size_t)64 * VP * 2);
  float*  part  = (float*)take((size_t)NB6 * 64 * 1024 * 4);
  float*  hbuf  = (float*)take((size_t)64 * 1024 * 4);
  float*  h1buf = (float*)take((size_t)64 * 1024 * 4);
  float*  rbuf  = (float*)take((size_t)512 * 1024 * 4);
  float*  rkbuf = (float*)take((size_t)512 * 1024 * 4);
  __bf16* kvbuf = (__bf16*)take((size_t)32768 * 2048 * 2);
  float*  qbuf  = (float*)take((size_t)64 * 1024 * 4);
  float*  avbuf = (float*)take((size_t)64 * 1024 * 4);
  float*  tmp1  = (float*)take((size_t)64 * 4096 * 4);
  float*  tmp2  = (float*)take((size_t)64 * 1024 * 4);

  // ---- FrozenHopfield retrieval ----
  k_mean<<<64, 256, 0, stream>>>(obs, xw);
  k_proj<<<16, 256, 0, stream>>>(xw, rand_proj, pw);
  k_rownorm<<<64, 256, 0, stream>>>(pw, pn);
  size_t simLds = (size_t)2 * 64 * SPAD * 2 + (64 + 512) * 4;
  k_sim<<<VP / 64, 512, simLds, stream>>>(pw, pn, word_embs, simw);
  k_rowmax<<<64, 1024, 0, stream>>>(simw, mrow);
  k_exp<<<64, 1024, 0, stream>>>(simw, mrow, probs, srow);
  size_t avLds = (size_t)1024 * TPAD * 2;
  k_av<<<NB6, 1024, avLds, stream>>>(probs, word_embs, part);
  k_merge<<<64, 1024, 0, stream>>>(part, srow, hbuf);

  // ---- TransfoXL layers ----
  k_posemb<<<512, 256, 0, stream>>>(rbuf);
  for (int l = 0; l < NL; ++l) {
    const float* mems_l = mems + (size_t)l * MEMN * 64 * 1024;
    const float* qkv_l  = qkv_w + (size_t)l * 1024 * 3072;
    // K,V for all 512 positions (cols 1024..3071 of qkv_w)
    k_gemm_cat<<<dim3(16, 256), 512, 0, stream>>>(mems_l, hbuf, qkv_l, 3072, 1024,
                                                  kvbuf, 2048, 1024);
    // q for the current token only (cols 0..1023)
    k_gemm<<<dim3(8, 1), 512, 0, stream>>>(hbuf, 1024, qkv_l, 3072, 0,
                                           qbuf, 1024, 64, 1024, nullptr, 0);
    // r_k = r @ r_w[l]
    k_gemm<<<dim3(8, 4), 512, 0, stream>>>(rbuf, 1024, r_w + (size_t)l * 1024 * 1024, 1024, 0,
                                           rkbuf, 1024, 512, 1024, nullptr, 0);
    k_attn<<<dim3(64, 16), 256, 0, stream>>>(qbuf, kvbuf, rkbuf,
                                             r_w_bias + (size_t)l * 1024,
                                             r_r_bias + (size_t)l * 1024, avbuf);
    // attn output projection
    k_gemm<<<dim3(8, 1), 512, 0, stream>>>(avbuf, 1024, o_w + (size_t)l * 1024 * 1024, 1024, 0,
                                           tmp2, 1024, 64, 1024, nullptr, 0);
    k_addln<<<64, 256, 0, stream>>>(hbuf, tmp2, ln1_s + (size_t)l * 1024,
                                    ln1_b + (size_t)l * 1024, h1buf);
    // FFN
    k_gemm<<<dim3(32, 1), 512, 0, stream>>>(h1buf, 1024, ff_w1 + (size_t)l * 1024 * 4096, 4096, 0,
                                            tmp1, 4096, 64, 1024,
                                            ff_b1 + (size_t)l * 4096, 2);
    k_gemm<<<dim3(8, 1), 512, 0, stream>>>(tmp1, 4096, ff_w2 + (size_t)l * 4096 * 1024, 1024, 0,
                                           tmp2, 1024, 64, 4096,
                                           ff_b2 + (size_t)l * 1024, 1);
    k_addln<<<64, 256, 0, stream>>>(h1buf, tmp2, ln2_s + (size_t)l * 1024,
                                    ln2_b + (size_t)l * 1024,
                                    (l == NL - 1) ? out : hbuf);
  }
}
